// GCN_LSTM_Model_47699906789559
// MI455X (gfx1250) — matrix operand order repaired and checked by reading, not compiled, not run on hardware
//
#include <hip/hip_runtime.h>
#include <hip/hip_bf16.h>
#include <math.h>

// ---------- constants for this problem ----------
#define BATCH 32
#define SEQ   24
#define NNOD  2000
#define NEDGE 16000
#define GCNF  8
#define HID   256
#define IFEAT 16000           // GCNF * NNOD
#define G4H   1024            // 4 * HID
#define ROWS  768             // BATCH * SEQ
#define OUTC  8000            // N_STEPS_OUT * NNOD

typedef _Float16 half8  __attribute__((ext_vector_type(8)));
typedef _Float16 half16 __attribute__((ext_vector_type(16)));
typedef float    float8 __attribute__((ext_vector_type(8)));

// ======================================================================
// Kernel 1: GCN per snapshot.  One workgroup per (b,s) graph.
// acc[n,f] = sum_{e: dst==n} dist_e * (x[src_e]*W[f] + b[f]);  out = tanh
// LDS accumulator 16000 floats = 64 KB (fits in 320 KB WGP LDS).
// Output written as f16 (A matrix of the big WMMA GEMM).
// ======================================================================
__global__ __launch_bounds__(256)
void k_gcn(const float* __restrict__ x, const int* __restrict__ ei,
           const float* __restrict__ ed, const float* __restrict__ Wg,
           const float* __restrict__ bg, _Float16* __restrict__ out16) {
    __shared__ float acc[IFEAT];
    const int snap = blockIdx.x;          // 0..767
    const int tid  = threadIdx.x;

    float w[GCNF], bb[GCNF];
#pragma unroll
    for (int f = 0; f < GCNF; ++f) { w[f] = Wg[f]; bb[f] = bg[f]; }

    for (int k = tid; k < IFEAT; k += 256) acc[k] = 0.0f;
    __syncthreads();

    const int*   src  = ei + (long)snap * 2 * NEDGE;
    const int*   dst  = src + NEDGE;
    const float* dist = ed + (long)snap * NEDGE;
    const float* xs   = x  + (long)snap * NNOD;

    for (int e = tid; e < NEDGE; e += 256) {
        const int   s  = src[e];
        const int   d  = dst[e];
        const float xv = xs[s];
        const float dv = dist[e];
        const int   base = d * GCNF;
#pragma unroll
        for (int f = 0; f < GCNF; ++f)
            atomicAdd(&acc[base + f], dv * (xv * w[f] + bb[f]));  // ds_add_f32
    }
    __syncthreads();

    _Float16* o = out16 + (long)snap * IFEAT;
    for (int k = tid; k < IFEAT; k += 256)
        o[k] = (_Float16)tanhf(acc[k]);
}

// ======================================================================
// Kernel 2: big GEMM  ih0[768,1024] = A16[768,16000] @ Wih0_16[1024,16000]^T
// Register-blocked: one wave32 computes a 32x32 macro-tile (2x2 WMMA tiles,
// 4 accumulators).  Per K-step: 2 A frags + 2 B frags -> 4 x
// v_wmma_f32_16x16x32_f16, i.e. 32 B/lane of L2 traffic per WMMA (2x better
// arithmetic intensity than 1-tile waves).  Both operands are L2-resident
// (24.6 MB + 32.8 MB << 192 MB L2).
// Fragment loads follow the CDNA5 16-bit A(16x32)/B(32x16) lane layouts:
//   A lane (m = lane&15): halves 0..7  -> K = k0 + (lane>=16?8:0)  + 0..7
//                         halves 8..15 -> K = k0+16+(lane>=16?8:0) + 0..7
//   B lane (n = lane&15): same K mapping as A
// Both operands are K-major in memory -> contiguous 16B b128 loads.
// ======================================================================
__device__ __forceinline__ half16 ld_afrag(const _Float16* p) {
    half8 lo = *(const half8*)(p);
    half8 hi = *(const half8*)(p + 16);
    return __builtin_shufflevector(lo, hi, 0,1,2,3,4,5,6,7,8,9,10,11,12,13,14,15);
}
__device__ __forceinline__ half16 ld_bfrag(const _Float16* p) {
    half8 lo = *(const half8*)(p);
    half8 hi = *(const half8*)(p + 16);
    return __builtin_shufflevector(lo, hi, 0,1,2,3,4,5,6,7,8,9,10,11,12,13,14,15);
}

__global__ __launch_bounds__(256)
void k_gemm_ih0(const _Float16* __restrict__ A,   // [768,16000] f16
                const _Float16* __restrict__ Bw,  // [1024,16000] f16
                float* __restrict__ C) {          // [768,1024] f32
    const int lane = threadIdx.x & 31;
    const int wave = threadIdx.x >> 5;
    const int tile = blockIdx.x * 8 + wave;       // 0..767 macro-tiles
    const int tm   = tile >> 5;                   // 0..23 (32-row macro tiles)
    const int tn   = tile & 31;                   // 0..31 (32-col macro tiles)
    const int l15  = lane & 15;
    const int hi   = lane >> 4;                   // 0 or 1

    // per-lane base rows for the two A fragments / two B fragments
    const long aBase0 = (long)(tm * 32 + l15)      * IFEAT;
    const long aBase1 = (long)(tm * 32 + 16 + l15) * IFEAT;
    const long bBase0 = (long)(tn * 32 + l15)      * IFEAT;
    const long bBase1 = (long)(tn * 32 + 16 + l15) * IFEAT;

    float8 acc00 = {0.f,0.f,0.f,0.f,0.f,0.f,0.f,0.f};
    float8 acc01 = acc00, acc10 = acc00, acc11 = acc00;

    const int aOff = hi * 8;     // A fragment K sub-offset per lane half
    const int bOff = hi * 8;     // B fragment K sub-offset per lane half

    for (int k0 = 0; k0 < IFEAT; k0 += 32) {
        const _Float16* ap0 = A  + aBase0 + k0 + aOff;
        const _Float16* ap1 = A  + aBase1 + k0 + aOff;
        const _Float16* bp0 = Bw + bBase0 + k0 + bOff;
        const _Float16* bp1 = Bw + bBase1 + k0 + bOff;

        // prefetch a few K-iterations ahead (global_prefetch_b8)
        __builtin_prefetch(ap0 + 128, 0, 1);
        __builtin_prefetch(ap1 + 128, 0, 1);
        __builtin_prefetch(bp0 + 128, 0, 1);
        __builtin_prefetch(bp1 + 128, 0, 1);

        half16 a0 = ld_afrag(ap0);
        half16 a1 = ld_afrag(ap1);
        half16 b0 = ld_bfrag(bp0);
        half16 b1 = ld_bfrag(bp1);

        acc00 = __builtin_amdgcn_wmma_f32_16x16x32_f16(
                    false, a0, false, b0, (short)0, acc00, false, false);
        acc01 = __builtin_amdgcn_wmma_f32_16x16x32_f16(
                    false, a0, false, b1, (short)0, acc01, false, false);
        acc10 = __builtin_amdgcn_wmma_f32_16x16x32_f16(
                    false, a1, false, b0, (short)0, acc10, false, false);
        acc11 = __builtin_amdgcn_wmma_f32_16x16x32_f16(
                    false, a1, false, b1, (short)0, acc11, false, false);
    }

    // C layout per tile: VGPR r -> M = r + 8*hi, N = lane&15
    const int m0 = tm * 32 + hi * 8;
    const int n0 = tn * 32 + l15;
#pragma unroll
    for (int r = 0; r < 8; ++r) {
        C[(long)(m0 + r)      * G4H + n0]      = acc00[r];
        C[(long)(m0 + r)      * G4H + n0 + 16] = acc01[r];
        C[(long)(m0 + 16 + r) * G4H + n0]      = acc10[r];
        C[(long)(m0 + 16 + r) * G4H + n0 + 16] = acc11[r];
    }
}

// ======================================================================
// Kernel 3: 2-layer LSTM recurrence. One workgroup per batch (32 WGs).
// Input gates for layer0 were precomputed by the GEMM (ih0).
// Weights pre-transposed to [K=256][4H=1024] so gate reads are coalesced.
// Thread `tid` owns hidden unit tid: gates at {tid, tid+256, tid+512, tid+768}.
// ======================================================================
__device__ __forceinline__ float sigmoidf_(float v) {
    return 1.0f / (1.0f + __expf(-v));
}

__global__ __launch_bounds__(256)
void k_lstm(const float* __restrict__ ih0,     // [768,1024]
            const float* __restrict__ Whh0T,   // [256,1024]
            const float* __restrict__ Wih1T,   // [256,1024]
            const float* __restrict__ Whh1T,   // [256,1024]
            const float* __restrict__ bias0,   // [1024] (bih0+bhh0)
            const float* __restrict__ bias1,   // [1024]
            float* __restrict__ last) {        // [32,256]  tanh(h2) at t=S-1
    __shared__ float h0[HID];
    __shared__ float h1[HID];
    const int b   = blockIdx.x;
    const int tid = threadIdx.x;

    float c0 = 0.0f, c1 = 0.0f;
    h0[tid] = 0.0f;
    h1[tid] = 0.0f;
    __syncthreads();

    const float b0i = bias0[tid],       b0f = bias0[tid + 256];
    const float b0g = bias0[tid + 512], b0o = bias0[tid + 768];
    const float b1i = bias1[tid],       b1f = bias1[tid + 256];
    const float b1g = bias1[tid + 512], b1o = bias1[tid + 768];

    for (int t = 0; t < SEQ; ++t) {
        // ---- layer 0 ----
        const float* ih = ih0 + (long)(b * SEQ + t) * G4H;
        float ai = ih[tid]       + b0i;
        float af = ih[tid + 256] + b0f;
        float ag = ih[tid + 512] + b0g;
        float ao = ih[tid + 768] + b0o;
        for (int k = 0; k < HID; ++k) {
            const float hk = h0[k];
            const float* wr = Whh0T + k * G4H;
            ai += wr[tid]       * hk;
            af += wr[tid + 256] * hk;
            ag += wr[tid + 512] * hk;
            ao += wr[tid + 768] * hk;
        }
        __syncthreads();                       // all reads of old h0 done
        c0 = sigmoidf_(af) * c0 + sigmoidf_(ai) * tanhf(ag);
        h0[tid] = sigmoidf_(ao) * tanhf(c0);
        __syncthreads();                       // new h0 visible

        // ---- layer 1 (input = new h0) ----
        float bi = b1i, bf = b1f, bg = b1g, bo = b1o;
        for (int k = 0; k < HID; ++k) {
            const float xk = h0[k];
            const float hk = h1[k];
            const float* wi = Wih1T + k * G4H;
            const float* wh = Whh1T + k * G4H;
            bi += wi[tid]       * xk + wh[tid]       * hk;
            bf += wi[tid + 256] * xk + wh[tid + 256] * hk;
            bg += wi[tid + 512] * xk + wh[tid + 512] * hk;
            bo += wi[tid + 768] * xk + wh[tid + 768] * hk;
        }
        __syncthreads();
        c1 = sigmoidf_(bf) * c1 + sigmoidf_(bi) * tanhf(bg);
        h1[tid] = sigmoidf_(bo) * tanhf(c1);
        __syncthreads();
    }
    last[b * HID + tid] = tanhf(h1[tid]);
}

// ======================================================================
// Kernel 4: FC  out[32,8000] = last @ fc_W^T + fc_b
// fc_W pre-transposed to [256,8000] -> coalesced weight loads;
// last (32x256 = 32 KB) in LDS, broadcast reads per (k,b).
// ======================================================================
__global__ __launch_bounds__(256)
void k_fc(const float* __restrict__ fcWT,   // [256,8000]
          const float* __restrict__ fcb,    // [8000]
          const float* __restrict__ last,   // [32,256]
          float* __restrict__ out) {        // [32,8000]
    __shared__ float ls[BATCH * HID];
    const int tid = threadIdx.x;
    for (int i = tid; i < BATCH * HID; i += 256) ls[i] = last[i];
    __syncthreads();

    const int col = blockIdx.x * 256 + tid;
    if (col < OUTC) {
        float acc[BATCH];
#pragma unroll
        for (int b = 0; b < BATCH; ++b) acc[b] = 0.0f;
        for (int k = 0; k < HID; ++k) {
            const float wv = fcWT[(long)k * OUTC + col];
#pragma unroll
            for (int b = 0; b < BATCH; ++b)
                acc[b] += wv * ls[b * HID + k];
        }
        const float bv = fcb[col];
#pragma unroll
        for (int b = 0; b < BATCH; ++b)
            out[(long)b * OUTC + col] = acc[b] + bv;
    }
}

// ======================================================================
// Prep kernels: f32->f16 convert, transpose, bias sum
// ======================================================================
__global__ void k_f32_to_f16(const float* __restrict__ src,
                             _Float16* __restrict__ dst, long n) {
    for (long i = blockIdx.x * (long)blockDim.x + threadIdx.x; i < n;
         i += (long)gridDim.x * blockDim.x)
        dst[i] = (_Float16)src[i];
}

__global__ void k_transpose(const float* __restrict__ src,
                            float* __restrict__ dst, int R, int C) {
    const long n = (long)R * C;
    for (long i = blockIdx.x * (long)blockDim.x + threadIdx.x; i < n;
         i += (long)gridDim.x * blockDim.x) {
        const long r = i / C;
        const long c = i - r * C;
        dst[c * (long)R + r] = src[i];
    }
}

__global__ void k_bias(const float* __restrict__ a, const float* __restrict__ b,
                       float* __restrict__ o, int n) {
    const int i = blockIdx.x * blockDim.x + threadIdx.x;
    if (i < n) o[i] = a[i] + b[i];
}

// ======================================================================
// Host launcher
// ======================================================================
extern "C" void kernel_launch(void* const* d_in, const int* in_sizes, int n_in,
                              void* d_out, int out_size, void* d_ws, size_t ws_size,
                              hipStream_t stream) {
    const float* x    = (const float*)d_in[0];
    const int*   ei   = (const int*)  d_in[1];
    const float* ed   = (const float*)d_in[2];
    const float* Wg   = (const float*)d_in[3];
    const float* bg   = (const float*)d_in[4];
    const float* Wih0 = (const float*)d_in[5];
    const float* Whh0 = (const float*)d_in[6];
    const float* bih0 = (const float*)d_in[7];
    const float* bhh0 = (const float*)d_in[8];
    const float* Wih1 = (const float*)d_in[9];
    const float* Whh1 = (const float*)d_in[10];
    const float* bih1 = (const float*)d_in[11];
    const float* bhh1 = (const float*)d_in[12];
    const float* fcW  = (const float*)d_in[13];
    const float* fcb  = (const float*)d_in[14];
    float* out = (float*)d_out;

    // workspace carving (256B aligned slices)
    char*  ws  = (char*)d_ws;
    size_t off = 0;
    auto take = [&](size_t bytes) -> char* {
        off = (off + 255) & ~(size_t)255;
        char* p = ws + off;
        off += bytes;
        return p;
    };
    _Float16* A16   = (_Float16*)take((size_t)ROWS * IFEAT * 2); // gcn_seq f16
    _Float16* B16   = (_Float16*)take((size_t)G4H * IFEAT * 2);  // Wih0 f16
    float*    ih0   = (float*)take((size_t)ROWS * G4H * 4);      // precomp gates
    float*    Whh0T = (float*)take((size_t)HID * G4H * 4);
    float*    Wih1T = (float*)take((size_t)HID * G4H * 4);
    float*    Whh1T = (float*)take((size_t)HID * G4H * 4);
    float*    bias0 = (float*)take((size_t)G4H * 4);
    float*    bias1 = (float*)take((size_t)G4H * 4);
    float*    lastv = (float*)take((size_t)BATCH * HID * 4);
    float*    fcWT  = (float*)take((size_t)HID * OUTC * 4);
    (void)ws_size; (void)n_in; (void)in_sizes; (void)out_size;

    // independent prep + GCN
    k_gcn<<<ROWS, 256, 0, stream>>>(x, ei, ed, Wg, bg, A16);
    k_f32_to_f16<<<1024, 256, 0, stream>>>(Wih0, B16, (long)G4H * IFEAT);
    k_transpose <<<256, 256, 0, stream>>>(Whh0, Whh0T, G4H, HID);
    k_transpose <<<256, 256, 0, stream>>>(Wih1, Wih1T, G4H, HID);
    k_transpose <<<256, 256, 0, stream>>>(Whh1, Whh1T, G4H, HID);
    k_transpose <<<512, 256, 0, stream>>>(fcW, fcWT, OUTC, HID);
    k_bias      <<<4, 256, 0, stream>>>(bih0, bhh0, bias0, G4H);
    k_bias      <<<4, 256, 0, stream>>>(bih1, bhh1, bias1, G4H);

    // big WMMA GEMM: 24x32 = 768 macro-tiles (32x32 each), 8 waves/block
    k_gemm_ih0<<<96, 256, 0, stream>>>(A16, B16, ih0);

    // sequential recurrence (one WG per batch), then FC
    k_lstm<<<BATCH, 256, 0, stream>>>(ih0, Whh0T, Wih1T, Whh1T, bias0, bias1, lastv);
    k_fc  <<<(OUTC + 255) / 256, 256, 0, stream>>>(fcWT, fcb, lastv, out);
}
